// MultiHeadAttention_19765439496418
// MI455X (gfx1250) — compile-verified, matrix operand order checked by reading, not run on hardware
//
#include <hip/hip_runtime.h>
#include <hip/hip_bf16.h>

// ---------------------------------------------------------------------------
// MHA forward for MI455X (gfx1250), wave32 + v_wmma_f32_16x16x32_bf16.
// Inputs (f32): v(8,1024,512), pad_mask(8,1,1,1024) bool, W1(512,2048),
//               b1(2048), W2(8192,512), b2(512)
// Outputs (f32, concat): out(8,1024,512) then attn(8,16,1024,1024)
// ---------------------------------------------------------------------------

typedef __bf16 bf16_t;
typedef __attribute__((ext_vector_type(16))) __bf16 v16bf;
typedef __attribute__((ext_vector_type(8)))  float  v8f;

#define WMMA_BF16(a, b, c) \
  __builtin_amdgcn_wmma_f32_16x16x32_bf16(false, (a), false, (b), (short)0, (c), false, false)

union FragU { v16bf v; uint4 q[2]; };

// A (16x32 MxK) or B (32x16 KxN, stored N-major) fragment from bf16 memory.
// ISA layout: lane L holds row/col (L&15); K chunks (L>>4)*8..+7 and +16..+23,
// packed 2-per-VGPR => vector elems 0..7 = first chunk, 8..15 = second chunk.
__device__ __forceinline__ v16bf load_frag_bf16(const bf16_t* base, int stride, int lane) {
  const int r  = lane & 15;
  const int kb = (lane >> 4) * 8;
  const bf16_t* p = base + (size_t)r * stride + kb;
  FragU f;
  f.q[0] = *(const uint4*)(p);
  f.q[1] = *(const uint4*)(p + 16);
  return f.v;
}

// Same fragment, but sourced from f32 row-major memory with on-the-fly cvt.
__device__ __forceinline__ v16bf load_frag_f32cvt(const float* base, int stride, int lane) {
  const int r  = lane & 15;
  const int kb = (lane >> 4) * 8;
  const float* p = base + (size_t)r * stride + kb;
  float4 x0 = *(const float4*)(p);
  float4 x1 = *(const float4*)(p + 4);
  float4 x2 = *(const float4*)(p + 16);
  float4 x3 = *(const float4*)(p + 20);
  v16bf o;
  o[0]  = (bf16_t)x0.x; o[1]  = (bf16_t)x0.y; o[2]  = (bf16_t)x0.z; o[3]  = (bf16_t)x0.w;
  o[4]  = (bf16_t)x1.x; o[5]  = (bf16_t)x1.y; o[6]  = (bf16_t)x1.z; o[7]  = (bf16_t)x1.w;
  o[8]  = (bf16_t)x2.x; o[9]  = (bf16_t)x2.y; o[10] = (bf16_t)x2.z; o[11] = (bf16_t)x2.w;
  o[12] = (bf16_t)x3.x; o[13] = (bf16_t)x3.y; o[14] = (bf16_t)x3.z; o[15] = (bf16_t)x3.w;
  return o;
}

// gfx1250 async copy: 16 bytes per lane, global -> LDS, tracked by ASYNCcnt.
__device__ __forceinline__ void async_copy_b128(unsigned int lds_addr, const void* gaddr) {
  asm volatile("global_load_async_to_lds_b128 %0, %1, off"
               :: "v"(lds_addr), "v"(gaddr)
               : "memory");
}
__device__ __forceinline__ void wait_asynccnt0() {
  asm volatile("s_wait_asynccnt 0x0" ::: "memory");
}

// --------------------------- stage 0: transposed bf16 copies ----------------
__global__ void k_transpose_bf16(const float* __restrict__ src, bf16_t* __restrict__ dst,
                                 int R, int C) {
  size_t idx = (size_t)blockIdx.x * blockDim.x + threadIdx.x;
  if (idx >= (size_t)R * C) return;
  int c = (int)(idx % C);
  int r = (int)(idx / C);
  dst[(size_t)c * R + r] = (bf16_t)src[idx];   // dst is C x R
}

__global__ void k_make_vT(const float* __restrict__ v, bf16_t* __restrict__ vT) {
  size_t idx = (size_t)blockIdx.x * blockDim.x + threadIdx.x;
  if (idx >= (size_t)8 * 1024 * 512) return;
  int c = (int)(idx & 511);
  size_t rs = idx >> 9;
  int s = (int)(rs & 1023);
  int b = (int)(rs >> 10);
  vT[((size_t)b * 512 + c) * 1024 + s] = (bf16_t)v[idx];  // vT[b] is 512 x 1024
}

// --------------------------- stage 1: kq = v @ W1 + b1 ----------------------
// M=8192 (b*s), K=512, N=2048. One wave computes a 16x64 output tile.
// Output scattered directly to k/q buffers, head-major bf16: [(b*16+h)][s][64].
__global__ void k_gemm1_kq(const float* __restrict__ V, const bf16_t* __restrict__ W1T,
                           const float* __restrict__ b1,
                           bf16_t* __restrict__ kbuf, bf16_t* __restrict__ qbuf) {
  const int lane = threadIdx.x & 31;
  const int gw   = (int)((blockIdx.x * blockDim.x + threadIdx.x) >> 5);
  const int NB = 2048 / 64;            // 32 column blocks
  const int mt = gw / NB;              // 0..511
  const int nb = gw % NB;
  const int row0 = mt * 16;
  const int n0   = nb * 64;

  v8f acc[4];
#pragma unroll
  for (int j = 0; j < 4; ++j)
#pragma unroll
    for (int i = 0; i < 8; ++i) acc[j][i] = 0.0f;

  for (int ks = 0; ks < 512; ks += 32) {
    v16bf a = load_frag_f32cvt(V + (size_t)row0 * 512 + ks, 512, lane);
#pragma unroll
    for (int j = 0; j < 4; ++j) {
      v16bf b = load_frag_bf16(W1T + (size_t)(n0 + j * 16) * 512 + ks, 512, lane);
      acc[j] = WMMA_BF16(a, b, acc[j]);
    }
  }

#pragma unroll
  for (int j = 0; j < 4; ++j) {
    const int col  = n0 + j * 16 + (lane & 15);
    const int h    = col >> 7;           // head
    const int r    = col & 127;          // 0..127 within head
    const float bias = b1[col];
    bf16_t* dst = (r < 64) ? kbuf : qbuf;
    const int d = r & 63;
#pragma unroll
    for (int i = 0; i < 8; ++i) {
      const int row = row0 + (lane >> 4) * 8 + i;
      const int b_  = row >> 10;
      const int s   = row & 1023;
      dst[(((size_t)(b_ * 16 + h)) * 1024 + s) * 64 + d] = (bf16_t)(acc[j][i] + bias);
    }
  }
}

// ------------------- stage 2: attn = softmax(mask(QK^T/8)) ------------------
// One wave per 16-row strip of one (b,h). All 8 waves of a block share the
// same (b,h): the 128 KB K-slice is staged ONCE per block into LDS with
// gfx1250 async global->LDS copies (ASYNCcnt), then both WMMA passes read
// B-fragments from LDS (ds_load_b128).
// Pass 1: online (flash) row max/sumexp. Pass 2: recompute, write probs.
__global__ void k_attn_softmax(const bf16_t* __restrict__ qbuf, const bf16_t* __restrict__ kbuf,
                               const unsigned char* __restrict__ mask,
                               float* __restrict__ attnOut) {
  extern __shared__ char smem[];                 // 128 KB dynamic LDS
  bf16_t* Ks = (bf16_t*)smem;                    // 1024 x 64 bf16

  const int lane = threadIdx.x & 31;
  const int tid  = threadIdx.x;                  // 0..255
  const int gw   = (int)((blockIdx.x * blockDim.x + threadIdx.x) >> 5);
  const int strip = gw & 63;                     // 16-row strip
  const int bh    = gw >> 6;                     // 0..127 (uniform per block)
  const int b_    = bh >> 4;

  const bf16_t* Q = qbuf + (size_t)bh * 1024 * 64 + (size_t)strip * 16 * 64;
  const bf16_t* K = kbuf + (size_t)bh * 1024 * 64;
  const unsigned char* mk = mask + (size_t)b_ * 1024;
  float* Srow = attnOut + (size_t)bh * 1024 * 1024 + (size_t)strip * 16 * 1024;

  // ---- cooperative async stage of K slice (1024x64 bf16 = 128 KB) ----
  {
    const unsigned int lds0 = (unsigned int)(uintptr_t)Ks;
    const char* gsrc = (const char*)K;
#pragma unroll 4
    for (int it = 0; it < 32; ++it) {            // 32 chunks * 256 thr * 16 B
      const unsigned int off = (unsigned int)(it * 256 + tid) * 16u;
      async_copy_b128(lds0 + off, gsrc + off);
    }
    wait_asynccnt0();                            // my copies landed in LDS
    __syncthreads();                             // everyone's copies visible
  }

  const v16bf a0 = load_frag_bf16(Q + 0,  64, lane);   // d = 0..31
  const v16bf a1 = load_frag_bf16(Q + 32, 64, lane);   // d = 32..63
  const int col = lane & 15;

  float m[8], l[8];
#pragma unroll
  for (int i = 0; i < 8; ++i) { m[i] = -3.0e38f; l[i] = 0.0f; }

  // ---- pass 1: online softmax statistics ----
  for (int nt = 0; nt < 64; ++nt) {
    v8f c;
#pragma unroll
    for (int i = 0; i < 8; ++i) c[i] = 0.0f;
    const bf16_t* Kt = Ks + (size_t)nt * 16 * 64;
    v16bf bA = load_frag_bf16(Kt + 0,  64, lane);
    v16bf bB = load_frag_bf16(Kt + 32, 64, lane);
    c = WMMA_BF16(a0, bA, c);
    c = WMMA_BF16(a1, bB, c);
    const bool msk = mk[nt * 16 + col] != 0;
#pragma unroll
    for (int i = 0; i < 8; ++i) {
      float s = msk ? -1000.0f : c[i] * 0.125f;
      float t = s;                           // row max across 16-lane group
#pragma unroll
      for (int d = 1; d < 16; d <<= 1) t = fmaxf(t, __shfl_xor(t, d, 32));
      const float mo = m[i];
      const float mn = fmaxf(mo, t);
      float e = __expf(s - mn);
#pragma unroll
      for (int d = 1; d < 16; d <<= 1) e += __shfl_xor(e, d, 32);
      l[i] = l[i] * __expf(mo - mn) + e;
      m[i] = mn;
    }
  }
  float rl[8];
#pragma unroll
  for (int i = 0; i < 8; ++i) rl[i] = 1.0f / l[i];

  // ---- pass 2: recompute scores, write normalized probabilities ----
  for (int nt = 0; nt < 64; ++nt) {
    v8f c;
#pragma unroll
    for (int i = 0; i < 8; ++i) c[i] = 0.0f;
    const bf16_t* Kt = Ks + (size_t)nt * 16 * 64;
    v16bf bA = load_frag_bf16(Kt + 0,  64, lane);
    v16bf bB = load_frag_bf16(Kt + 32, 64, lane);
    c = WMMA_BF16(a0, bA, c);
    c = WMMA_BF16(a1, bB, c);
    const bool msk = mk[nt * 16 + col] != 0;
#pragma unroll
    for (int i = 0; i < 8; ++i) {
      const float s = msk ? -1000.0f : c[i] * 0.125f;
      const float p = __expf(s - m[i]) * rl[i];
      const int r = (lane >> 4) * 8 + i;
      Srow[(size_t)r * 1024 + nt * 16 + col] = p;
    }
  }
}

// --------------------- stage 3: heads = attn @ v ----------------------------
// Per (b,h): (1024x1024 f32) @ (1024x512). A cvt'd f32->bf16 on load, B = vT bf16.
// Output bf16 laid out (b, s, h*512+c) as rows of an 8192x8192 matrix.
__global__ void k_attn_v(const float* __restrict__ attnOut, const bf16_t* __restrict__ vT,
                         bf16_t* __restrict__ heads) {
  const int lane = threadIdx.x & 31;
  const int gw   = (int)((blockIdx.x * blockDim.x + threadIdx.x) >> 5);
  const int nb = gw & 7;               // 8 x 64-col blocks
  const int mt = (gw >> 3) & 63;       // 64 x 16-row tiles
  const int bh = gw >> 9;              // 0..127
  const int b_ = bh >> 4, h = bh & 15;

  const float*  A = attnOut + (size_t)bh * 1024 * 1024 + (size_t)mt * 16 * 1024;
  const bf16_t* B = vT + (size_t)b_ * 512 * 1024;

  v8f acc[4];
#pragma unroll
  for (int j = 0; j < 4; ++j)
#pragma unroll
    for (int i = 0; i < 8; ++i) acc[j][i] = 0.0f;

  for (int ks = 0; ks < 1024; ks += 32) {
    v16bf a = load_frag_f32cvt(A + ks, 1024, lane);
#pragma unroll
    for (int j = 0; j < 4; ++j) {
      v16bf b = load_frag_bf16(B + (size_t)(nb * 64 + j * 16) * 1024 + ks, 1024, lane);
      acc[j] = WMMA_BF16(a, b, acc[j]);
    }
  }

#pragma unroll
  for (int j = 0; j < 4; ++j) {
    const int colC = h * 512 + nb * 64 + j * 16 + (lane & 15);
#pragma unroll
    for (int i = 0; i < 8; ++i) {
      const int row = mt * 16 + (lane >> 4) * 8 + i;
      heads[((size_t)b_ * 1024 + row) * 8192 + colC] = (bf16_t)acc[j][i];
    }
  }
}

// --------------------- stage 4: out = heads @ W2 + b2 -----------------------
// M=8192, K=8192, N=512.
__global__ void k_gemm2(const bf16_t* __restrict__ heads, const bf16_t* __restrict__ W2T,
                        const float* __restrict__ b2, float* __restrict__ out) {
  const int lane = threadIdx.x & 31;
  const int gw   = (int)((blockIdx.x * blockDim.x + threadIdx.x) >> 5);
  const int nb = gw & 7;               // 8 x 64-col blocks
  const int mt = gw >> 3;              // 0..511
  const size_t arow = (size_t)mt * 16 * 8192;

  v8f acc[4];
#pragma unroll
  for (int j = 0; j < 4; ++j)
#pragma unroll
    for (int i = 0; i < 8; ++i) acc[j][i] = 0.0f;

  for (int ks = 0; ks < 8192; ks += 32) {
    v16bf a = load_frag_bf16(heads + arow + ks, 8192, lane);
#pragma unroll
    for (int j = 0; j < 4; ++j) {
      v16bf b = load_frag_bf16(W2T + (size_t)(nb * 64 + j * 16) * 8192 + ks, 8192, lane);
      acc[j] = WMMA_BF16(a, b, acc[j]);
    }
  }

#pragma unroll
  for (int j = 0; j < 4; ++j) {
    const int col = nb * 64 + j * 16 + (lane & 15);
    const float bias = b2[col];
#pragma unroll
    for (int i = 0; i < 8; ++i) {
      const int row = mt * 16 + (lane >> 4) * 8 + i;
      out[(size_t)row * 512 + col] = acc[j][i] + bias;
    }
  }
}

// ---------------------------------------------------------------------------
extern "C" void kernel_launch(void* const* d_in, const int* in_sizes, int n_in,
                              void* d_out, int out_size, void* d_ws, size_t ws_size,
                              hipStream_t stream) {
  const float*         v    = (const float*)d_in[0];
  const unsigned char* mask = (const unsigned char*)d_in[1];   // bool array
  const float*         W1   = (const float*)d_in[2];
  const float*         b1   = (const float*)d_in[3];
  const float*         W2   = (const float*)d_in[4];
  const float*         b2   = (const float*)d_in[5];

  float* out     = (float*)d_out;                 // 8*1024*512
  float* attnOut = out + (size_t)8 * 1024 * 512;  // 8*16*1024*1024

  // workspace layout (bytes)
  char* ws = (char*)d_ws;
  bf16_t* kbuf  = (bf16_t*)(ws);                        // 16 MiB
  bf16_t* qbuf  = (bf16_t*)(ws + 16777216);             // 16 MiB
  bf16_t* W1T   = (bf16_t*)(ws + 33554432);             //  2 MiB  (2048 x 512)
  bf16_t* W2T   = (bf16_t*)(ws + 35651584);             //  8 MiB  (512 x 8192)
  bf16_t* vT    = (bf16_t*)(ws + 44040192);             //  8 MiB  (8 x 512 x 1024)
  bf16_t* heads = (bf16_t*)(ws + 52428800);             // 128 MiB (8192 x 8192)

  // stage 0: transposed bf16 operand copies
  k_transpose_bf16<<<(512 * 2048 + 255) / 256, 256, 0, stream>>>(W1, W1T, 512, 2048);
  k_transpose_bf16<<<(8192 * 512 + 255) / 256, 256, 0, stream>>>(W2, W2T, 8192, 512);
  k_make_vT<<<(8 * 1024 * 512 + 255) / 256, 256, 0, stream>>>(v, vT);

  // stage 1: kq projection (512 Mtiles * 32 Nblocks waves, 8 waves/block)
  k_gemm1_kq<<<2048, 256, 0, stream>>>(v, W1T, b1, kbuf, qbuf);

  // stage 2: scores + softmax; 128 KB dynamic LDS per block for the K slice
  k_attn_softmax<<<1024, 256, 131072, stream>>>(qbuf, kbuf, mask, attnOut);

  // stage 3: attn @ v (8*16*64*8 waves)
  k_attn_v<<<8192, 256, 0, stream>>>(attnOut, vT, heads);

  // stage 4: output projection (512*8 waves)
  k_gemm2<<<512, 256, 0, stream>>>(heads, W2T, b2, out);
}
